// MLAS_73289321939109
// MI455X (gfx1250) — matrix operand order repaired and checked by reading, not compiled, not run on hardware
//
#include <hip/hip_runtime.h>

// ---------------------------------------------------------------------------
// Feature detection (compile-only environment; guarded so we always compile)
// ---------------------------------------------------------------------------
#if __has_builtin(__builtin_amdgcn_tensor_load_to_lds) && \
    __has_builtin(__builtin_amdgcn_s_wait_tensorcnt)
#define HAVE_TDM 1
#else
#define HAVE_TDM 0
#endif
#if __has_include(<hip/amd_detail/amd_gfx1250_TDM.h>)
#define TDM_6ARG 1
#else
#define TDM_6ARG 0
#endif

typedef __attribute__((ext_vector_type(16))) _Float16 v16h;
typedef __attribute__((ext_vector_type(8)))  _Float16 v8h;
typedef __attribute__((ext_vector_type(8)))  float    v8f;

#define N_NODES 4096
#define IN_DIM  1024

#if HAVE_TDM
typedef __attribute__((ext_vector_type(4))) unsigned int u32x4;
typedef __attribute__((ext_vector_type(4))) int          i32x4;
typedef __attribute__((ext_vector_type(8))) int          i32x8;

__device__ __forceinline__ unsigned lds_addr_of(void* p) {
    // generic -> LDS address space, then to byte offset within the WG's LDS
    return (unsigned)(unsigned long long)(__attribute__((address_space(3))) void*)p;
}

// TDM 2D tile load: 128 rows x 32 halves from a row-major f16 matrix
// (row stride = STRIDE halves) into LDS at lds_byte_off.
// D# padding: 4 dwords after every 16 dwords -> LDS row stride 40 halves.
template <unsigned STRIDE>
__device__ __forceinline__ void tdm_load_tile_f16(const _Float16* gsrc,
                                                  unsigned lds_byte_off) {
    unsigned long long ga = (unsigned long long)gsrc;
    u32x4 g0;
    g0.x = 1u;                                                 // count=1 (valid user D#)
    g0.y = lds_byte_off;                                       // lds_addr (bytes)
    g0.z = (unsigned)ga;                                       // global_addr[31:0]
    g0.w = (unsigned)((ga >> 32) & 0x01ffffffu) | (2u << 30);  // addr[56:32] | type=2
    i32x8 g1;
    g1[0] = (int)((1u << 16) |          // data_size = 2 bytes
                  (1u << 20) |          // pad_enable
                  (3u << 22) |          // pad_interval = 16 dwords (64B)
                  (3u << 25));          // pad_amount   = 4 dwords (16B)
    g1[1] = (int)(32u << 16);           // tensor_dim0 = 32 (low 16 bits)
    g1[2] = (int)(128u << 16);          // tensor_dim0 hi=0 | tensor_dim1 = 128
    g1[3] = (int)(32u << 16);           // tensor_dim1 hi=0 | tile_dim0 = 32
    g1[4] = (int)128u;                  // tile_dim1 = 128 | tile_dim2 = 0
    g1[5] = (int)STRIDE;                // tensor_dim0_stride[31:0]
    g1[6] = 0;                          // stride0[47:32] | stride1[15:0]
    g1[7] = 0;                          // stride1[47:16]
    i32x4 z4 = (i32x4)0;
#if TDM_6ARG
    i32x8 z8 = (i32x8)0;
    __builtin_amdgcn_tensor_load_to_lds(g0, g1, z4, z4, z8, 0);
#else
    __builtin_amdgcn_tensor_load_to_lds(g0, g1, z4, z4, 0);
#endif
}
#endif  // HAVE_TDM

// ---------------------------------------------------------------------------
// f32 -> f16 conversion
// ---------------------------------------------------------------------------
__launch_bounds__(256)
__global__ void cvt_f32_to_f16(const float* __restrict__ in,
                               _Float16* __restrict__ out, int n) {
    int i = blockIdx.x * 256 + threadIdx.x;
    if (i < n) out[i] = (_Float16)in[i];
}

// ---------------------------------------------------------------------------
// Tiled WMMA GEMM: C[M x N] = A[M x K] * B, f16 in, f32 accumulate.
// All leading dimensions / K are compile-time so addresses fold to immediates.
//   B_KXN = false : B source is [N x K] row-major (NT gemm) -> TDM/direct stage
//   B_KXN = true  : B source is [K x N] row-major (NN gemm) -> transpose stage
// 256 threads (8 wave32), 128x128 C tile, K-step 32, double-buffered LDS.
// Each wave: 64x32 region = 4x2 v_wmma_f32_16x16x32_f16 accumulators.
// ---------------------------------------------------------------------------
template <int LDA, int LDB, int LDC, int KDIM, bool B_KXN, bool DO_RELU, bool WRITE_F32>
__launch_bounds__(256)
__global__ void gemm_wmma_f16(const _Float16* __restrict__ A,
                              const _Float16* __restrict__ B,
                              _Float16* __restrict__ Cf16,
                              float* __restrict__ Cf32,
                              float scale) {
    constexpr int LDST   = 40;          // padded LDS row stride in halves
    constexpr int TILE_H = 128 * LDST;  // halves per tile buffer (10 KB)
    __shared__ __align__(16) _Float16 smem[4 * TILE_H];  // A0 A1 B0 B1 = 40 KB

    const int tid  = threadIdx.x;
    const int lane = tid & 31;
    const int wid  = tid >> 5;
    const int wm   = (wid >> 2) * 64;   // wave row offset in C tile
    const int wn   = (wid & 3) * 32;    // wave col offset in C tile
    const int lr   = lane & 15;
    const int lh   = lane >> 4;

    const long tile_i = (long)blockIdx.y * 128;
    const long tile_j = (long)blockIdx.x * 128;

    // manual-staging geometry
    const int arow = tid >> 2;          // 0..63, chunk (tid&3)*8; rows arow, arow+64
    const int ac8  = (tid & 3) * 8;
    const int bkr0 = tid >> 4;          // 0..15 (B_KXN k-row); second chunk +16
    const int bc8  = (tid & 15) * 8;

    v8h tA[2], tB[2];

    auto issue_stage = [&](int b, int kk) {
#if HAVE_TDM
        (void)tA;
        if (wid == 0) {
            tdm_load_tile_f16<(unsigned)LDA>(
                A + tile_i * LDA + kk,
                lds_addr_of(smem) + (unsigned)(b * TILE_H) * 2u);
            if constexpr (!B_KXN)
                tdm_load_tile_f16<(unsigned)LDB>(
                    B + tile_j * LDB + kk,
                    lds_addr_of(smem) + (unsigned)((2 + b) * TILE_H) * 2u);
        }
        if constexpr (B_KXN) {
            tB[0] = *(const v8h*)(B + (long)(kk + bkr0) * LDB + tile_j + bc8);
            tB[1] = *(const v8h*)(B + (long)(kk + bkr0 + 16) * LDB + tile_j + bc8);
        } else {
            (void)tB;
        }
#else
        tA[0] = *(const v8h*)(A + (tile_i + arow) * LDA + kk + ac8);
        tA[1] = *(const v8h*)(A + (tile_i + arow + 64) * LDA + kk + ac8);
        if constexpr (!B_KXN) {
            tB[0] = *(const v8h*)(B + (tile_j + arow) * LDB + kk + ac8);
            tB[1] = *(const v8h*)(B + (tile_j + arow + 64) * LDB + kk + ac8);
        } else {
            tB[0] = *(const v8h*)(B + (long)(kk + bkr0) * LDB + tile_j + bc8);
            tB[1] = *(const v8h*)(B + (long)(kk + bkr0 + 16) * LDB + tile_j + bc8);
        }
        if (kk + 32 < KDIM)  // next slice into caches (global_prefetch_b8)
            __builtin_prefetch(A + (tile_i + arow) * LDA + kk + 32, 0, 0);
#endif
    };

    auto commit_stage = [&](int b) {
        _Float16* Asb = smem + b * TILE_H;
        _Float16* Bsb = smem + (2 + b) * TILE_H;
#if HAVE_TDM
        (void)Asb;
        if constexpr (B_KXN) {
            #pragma unroll
            for (int t = 0; t < 8; ++t) {
                Bsb[(bc8 + t) * LDST + bkr0]      = tB[0][t];
                Bsb[(bc8 + t) * LDST + bkr0 + 16] = tB[1][t];
            }
        } else {
            (void)Bsb;
        }
        if (wid == 0) __builtin_amdgcn_s_wait_tensorcnt(0);
#else
        *(v8h*)(Asb + arow * LDST + ac8)        = tA[0];
        *(v8h*)(Asb + (arow + 64) * LDST + ac8) = tA[1];
        if constexpr (!B_KXN) {
            *(v8h*)(Bsb + arow * LDST + ac8)        = tB[0];
            *(v8h*)(Bsb + (arow + 64) * LDST + ac8) = tB[1];
        } else {
            #pragma unroll
            for (int t = 0; t < 8; ++t) {
                Bsb[(bc8 + t) * LDST + bkr0]      = tB[0][t];
                Bsb[(bc8 + t) * LDST + bkr0 + 16] = tB[1][t];
            }
        }
#endif
    };

    v8f acc[4][2] = {};

    auto do_tile = [&](int b, int kk, bool more) {
        __syncthreads();                       // tile `b` published
        if (more) issue_stage(b ^ 1, kk + 32);

        const _Float16* Asb = smem + b * TILE_H;
        const _Float16* Bsb = smem + (2 + b) * TILE_H;

        // B fragments: lane<16: K 0-7 & 16-23; lane>=16: K 8-15 & 24-31
        v16h bf[2];
        #pragma unroll
        for (int nt = 0; nt < 2; ++nt) {
            const _Float16* bp = Bsb + (wn + nt * 16 + lr) * LDST + lh * 8;
            v8h blo = *(const v8h*)bp;
            v8h bhi = *(const v8h*)(bp + 16);
            #pragma unroll
            for (int t = 0; t < 8; ++t) { bf[nt][t] = blo[t]; bf[nt][t + 8] = bhi[t]; }
        }
        // A fragments: lane<16: K 0-7 & 16-23; lane>=16: K 8-15 & 24-31
        v16h af[4];
        #pragma unroll
        for (int mt = 0; mt < 4; ++mt) {
            const _Float16* ap = Asb + (wm + mt * 16 + lr) * LDST + lh * 8;
            v8h alo = *(const v8h*)ap;
            v8h ahi = *(const v8h*)(ap + 16);
            #pragma unroll
            for (int t = 0; t < 8; ++t) { af[mt][t] = alo[t]; af[mt][t + 8] = ahi[t]; }
        }
        #pragma unroll
        for (int mt = 0; mt < 4; ++mt)
            #pragma unroll
            for (int nt = 0; nt < 2; ++nt)
                acc[mt][nt] = __builtin_amdgcn_wmma_f32_16x16x32_f16(
                    false, af[mt], false, bf[nt],
                    (short)0, acc[mt][nt], false, false);

        if (more) commit_stage(b ^ 1);
    };

    // prologue: stage tile 0, then K loop unrolled x2 (KDIM % 64 == 0) so the
    // double-buffer index and LDS bases are compile-time constants.
    issue_stage(0, 0);
    commit_stage(0);
    for (int k0 = 0; k0 < KDIM; k0 += 64) {
        do_tile(0, k0, true);
        do_tile(1, k0 + 32, k0 + 64 < KDIM);
    }

    // ---- store C (lane: N = lr, M = lh*8 + r per accumulator VGPR r) ----
    // One base address per 16x16 tile; row steps are immediate offsets.
    #pragma unroll
    for (int mt = 0; mt < 4; ++mt) {
        #pragma unroll
        for (int nt = 0; nt < 2; ++nt) {
            const long row0 = tile_i + wm + mt * 16 + lh * 8;
            const long col  = tile_j + wn + nt * 16 + lr;
            _Float16* cp = Cf16 + row0 * LDC + col;
            float*    fp = Cf32 + row0 * LDC + col;  // only used if WRITE_F32
            #pragma unroll
            for (int r = 0; r < 8; ++r) {
                float v = acc[mt][nt][r] * scale;
                if (DO_RELU) v = fmaxf(v, 0.0f);
                cp[r * LDC] = (_Float16)v;
                if constexpr (WRITE_F32) fp[r * LDC] = v;
            }
        }
    }
}

// ---------------------------------------------------------------------------
// Row-wise softmax over a [N_NODES x N_NODES] f16 matrix, in place.
// ---------------------------------------------------------------------------
__launch_bounds__(256)
__global__ void softmax_rows(_Float16* __restrict__ S) {
    const int row = blockIdx.x;
    _Float16* rp  = S + (long)row * N_NODES;
    const int tid = threadIdx.x;

    float vals[16];
    #pragma unroll
    for (int c = 0; c < 2; ++c) {
        v8h v = *(const v8h*)(rp + (tid + c * 256) * 8);
        #pragma unroll
        for (int t = 0; t < 8; ++t) vals[c * 8 + t] = (float)v[t];
    }

    float m = -1e30f;
    #pragma unroll
    for (int i = 0; i < 16; ++i) m = fmaxf(m, vals[i]);
    #pragma unroll
    for (int off = 16; off >= 1; off >>= 1)
        m = fmaxf(m, __shfl_xor(m, off, 32));
    __shared__ float redm[8];
    __shared__ float reds[8];
    if ((tid & 31) == 0) redm[tid >> 5] = m;
    __syncthreads();
    float bm = redm[0];
    #pragma unroll
    for (int w = 1; w < 8; ++w) bm = fmaxf(bm, redm[w]);

    float s = 0.0f;
    #pragma unroll
    for (int i = 0; i < 16; ++i) { vals[i] = __expf(vals[i] - bm); s += vals[i]; }
    #pragma unroll
    for (int off = 16; off >= 1; off >>= 1)
        s += __shfl_xor(s, off, 32);
    if ((tid & 31) == 0) reds[tid >> 5] = s;
    __syncthreads();
    float total = 0.0f;
    #pragma unroll
    for (int w = 0; w < 8; ++w) total += reds[w];
    const float inv = 1.0f / total;

    #pragma unroll
    for (int c = 0; c < 2; ++c) {
        v8h o;
        #pragma unroll
        for (int t = 0; t < 8; ++t) o[t] = (_Float16)(vals[c * 8 + t] * inv);
        *(v8h*)(rp + (tid + c * 256) * 8) = o;
    }
}

// ---------------------------------------------------------------------------
// Host-side launcher
// ---------------------------------------------------------------------------
extern "C" void kernel_launch(void* const* d_in, const int* in_sizes, int n_in,
                              void* d_out, int out_size, void* d_ws, size_t ws_size,
                              hipStream_t stream) {
    (void)in_sizes; (void)n_in; (void)out_size; (void)ws_size;
    const float* Hin = (const float*)d_in[0];
    float* out       = (float*)d_out;

    char* ws = (char*)d_ws;
    const size_t hbytes = (size_t)N_NODES * IN_DIM * sizeof(_Float16);  // 8 MB
    _Float16* H0 = (_Float16*)ws;
    _Float16* H1 = (_Float16*)(ws + hbytes);
    _Float16* S  = (_Float16*)(ws + 2 * hbytes);  // 4096x4096 f16 = 32 MB

    const int n_elems = N_NODES * IN_DIM;
    cvt_f32_to_f16<<<(n_elems + 255) / 256, 256, 0, stream>>>(Hin, H0, n_elems);

    const float scale = 0.03125f;  // 1/sqrt(1024)
    _Float16* cur = H0;
    _Float16* nxt = H1;

    for (int layer = 0; layer < 3; ++layer) {
        // S = (cur * cur^T) * scale    [4096 x 4096]
        dim3 g1(N_NODES / 128, N_NODES / 128);
        gemm_wmma_f16<IN_DIM, IN_DIM, N_NODES, IN_DIM, false, false, false>
            <<<g1, 256, 0, stream>>>(cur, cur, S, nullptr, scale);

        // P = softmax_rows(S), in place
        softmax_rows<<<N_NODES, 256, 0, stream>>>(S);

        // nxt = relu(P * cur)          [4096 x 1024]
        dim3 g2(IN_DIM / 128, N_NODES / 128);
        if (layer == 2) {
            gemm_wmma_f16<N_NODES, IN_DIM, IN_DIM, N_NODES, true, true, true>
                <<<g2, 256, 0, stream>>>(S, cur, nxt, out, 1.0f);
        } else {
            gemm_wmma_f16<N_NODES, IN_DIM, IN_DIM, N_NODES, true, true, false>
                <<<g2, 256, 0, stream>>>(S, cur, nxt, nullptr, 1.0f);
        }

        _Float16* t = cur; cur = nxt; nxt = t;
    }
}
